// DAC_block_23888608101170
// MI455X (gfx1250) — matrix operand order repaired and checked by reading, not compiled, not run on hardware
//
#include <hip/hip_runtime.h>

// ---------------- model dims ----------------
#define NTOK   32768      // BATCH*SEQLEN
#define LSEQ   4096
#define NBATCH 8
#define DMODEL 768
#define DINNER 1536
#define DTRANK 48
#define DSTATE 32
#define NHID   2048
#define ABINS  256

typedef unsigned short u16;
typedef __attribute__((ext_vector_type(16))) __bf16 bf16x16;
typedef __attribute__((ext_vector_type(8)))  float  f32x8;

union AFrag { bf16x16 v; unsigned int u[8]; };

__device__ __forceinline__ u16 f2bf(float f) {
  unsigned int u = __float_as_uint(f);
  unsigned int r = u + 0x7FFFu + ((u >> 16) & 1u);   // RNE
  return (u16)(r >> 16);
}
__device__ __forceinline__ float bf2f(u16 h) {
  return __uint_as_float(((unsigned int)h) << 16);
}
// low 32 bits of a generic LDS pointer = LDS byte offset
__device__ __forceinline__ unsigned ldsAddr(const void* p) {
  return (unsigned)(unsigned long long)p;
}

// ---------------- elementwise kernels ----------------
__global__ void k_f2bf(const float* __restrict__ in, u16* __restrict__ out, long n) {
  long i = (long)blockIdx.x * blockDim.x + threadIdx.x;
  if (i < n) out[i] = f2bf(in[i]);
}

// convert + transpose weight: in [K,N] f32 -> out [N,K] bf16
__global__ void k_f2bf_t(const float* __restrict__ in, u16* __restrict__ out,
                         int K, int N) {
  long i = (long)blockIdx.x * blockDim.x + threadIdx.x;
  if (i >= (long)K * N) return;
  int n = (int)(i / K);
  int k = (int)(i % K);
  out[i] = f2bf(in[(long)k * N + n]);
}

// W_dt [48,1536] -> transposed padded bf16 [1536,64] (k 48..63 zero)
__global__ void k_pad_wdt_t(const float* __restrict__ in, u16* __restrict__ out) {
  long i = (long)blockIdx.x * blockDim.x + threadIdx.x;
  if (i >= (long)DINNER * 64) return;
  int n = (int)(i >> 6);
  int k = (int)(i & 63);
  out[i] = (k < DTRANK) ? f2bf(in[(long)k * DINNER + n]) : (u16)0;
}

// extract dt columns of proj [NTOK,112] -> bf16 [NTOK,64] (cols 48..63 zero)
__global__ void k_extract_dt(const float* __restrict__ proj, u16* __restrict__ out) {
  long i = (long)blockIdx.x * blockDim.x + threadIdx.x;
  if (i >= (long)NTOK * 64) return;
  int k = (int)(i & 63);
  long row = i >> 6;
  out[i] = (k < DTRANK) ? f2bf(proj[row * 112 + k]) : (u16)0;
}

// causal depthwise conv1d + bias + SiLU; writes f32 and bf16 copies
__global__ void k_conv(const float* __restrict__ xc, const float* __restrict__ cw,
                       const float* __restrict__ cb, float* __restrict__ xcs,
                       u16* __restrict__ xcs_bf) {
  long i = (long)blockIdx.x * blockDim.x + threadIdx.x;
  if (i >= (long)NTOK * DINNER) return;
  int d = (int)(i % DINNER);
  long t = i / DINNER;
  int l = (int)(t % LSEQ);
  float s = cb[d];
#pragma unroll
  for (int j = 0; j < 4; ++j) {
    int dl = l - 3 + j;
    if (dl >= 0) s += cw[d * 4 + j] * xc[(t - 3 + j) * DINNER + d];
  }
  float r = s / (1.f + __expf(-s));
  xcs[i] = r;
  xcs_bf[i] = f2bf(r);
}

// y_gated = (y_scan + xcs*D) * silu(z)  -> bf16
__global__ void k_gate(const float* __restrict__ y, const float* __restrict__ xcs,
                       const float* __restrict__ Dv, const u16* __restrict__ z,
                       u16* __restrict__ yg) {
  long i = (long)blockIdx.x * blockDim.x + threadIdx.x;
  if (i >= (long)NTOK * DINNER) return;
  int d = (int)(i % DINNER);
  float zv = bf2f(z[i]);
  float g = zv / (1.f + __expf(-zv));
  yg[i] = f2bf((y[i] + xcs[i] * Dv[d]) * g);
}

// ---------------- selective scan ----------------
// 4 lanes per (b,d) channel, 8 states per lane. Block = 64 channels (one batch).
__global__ __launch_bounds__(256)
void k_scan(const float* __restrict__ dtf, const float* __restrict__ proj,
            const float* __restrict__ xcs, const float* __restrict__ A_log,
            float* __restrict__ y) {
  __shared__ float bc[64];
  const int tid = threadIdx.x;
  const long g = (long)blockIdx.x * 256 + tid;
  const int c = (int)(g >> 2);   // channel index 0..12287
  const int q = (int)(g & 3);    // state quarter
  const int b = c / DINNER;
  const int d = c % DINNER;
  float a[8], h[8];
#pragma unroll
  for (int j = 0; j < 8; ++j) {
    a[j] = -__expf(A_log[(long)d * DSTATE + q * 8 + j]);
    h[j] = 0.f;
  }
  for (int t = 0; t < LSEQ; ++t) {
    long row = (long)b * LSEQ + t;
    if (tid < 64) bc[tid] = proj[row * 112 + DTRANK + tid];  // B[32] then C[32]
    __syncthreads();
    float dtv = dtf[row * DINNER + d];
    float xv  = xcs[row * DINNER + d];
    float dx  = dtv * xv;
    float acc = 0.f;
#pragma unroll
    for (int j = 0; j < 8; ++j) {
      int n = q * 8 + j;
      float dA = __expf(dtv * a[j]);
      h[j] = dA * h[j] + dx * bc[n];
      acc += h[j] * bc[32 + n];
    }
    acc += __shfl_xor(acc, 1, 32);
    acc += __shfl_xor(acc, 2, 32);
    if (q == 0) y[row * DINNER + d] = acc;
    __syncthreads();
  }
}

// ---------------- bf16 WMMA GEMM, async-LDS pipeline, fused epilogues ----------------
// C = A[M,K] * Bt[N,K]^T : A row-major [M,K], B pre-transposed [N,K] (both bf16/u16).
// 256 threads = 8 waves, block tile 128(M) x 128(N), K-tile 64; wave tile 32x64
// = 2x4 accumulators -> 16 WMMAs per K-tile per wave.
// Tiles move global->LDS with GLOBAL_LOAD_ASYNC_TO_LDS_B128 (no VGPR staging, no
// spills), double-buffered in LDS (64 KB), synced with s_wait_asynccnt + one barrier.
// GUARD=true only for N not divisible by 128 (xproj, N=112): invalid B rows are
// zero-filled once and never written by async copies.
// EPI 0: f32 store (C0)            EPI 1: split: col<splitN -> f32 C0, else bf16 C1
// EPI 2: softplus(acc+bias) f32    EPI 3: bf16 store (C1)
// EPI 4: leaky_relu(acc+bias) bf16 EPI 5: acc+bias f32
template <int EPI, bool GUARD>
__global__ __launch_bounds__(256)
void gemm_bf16(const u16* __restrict__ A, const u16* __restrict__ Bt,
               int M, int N, int K, int lda, int ldbT,
               float* __restrict__ C0, u16* __restrict__ C1,
               const float* __restrict__ bias, int splitN) {
  __shared__ __align__(16) u16 As[2][128 * 64];   // [m][k], 16 KB each
  __shared__ __align__(16) u16 Bs[2][128 * 64];   // [n][k], 16 KB each
  const int tid  = threadIdx.x;
  const int lane = tid & 31;
  const int wave = tid >> 5;
  const int wm = wave & 3;       // 4 waves along M
  const int wn = wave >> 2;      // 2 waves along N
  const int m0 = blockIdx.y * 128;
  const int n0 = blockIdx.x * 128;
  const int half = lane >> 4;
  const int l16  = lane & 15;

  // copy roles: row cr (0..127), 32-wide k half ch
  const int cr = tid >> 1, ch = tid & 1;
  const u16* aptr = A  + (size_t)(m0 + cr) * lda  + ch * 32;
  const u16* bptr = Bt + (size_t)(n0 + cr) * ldbT + ch * 32;
  const bool bvalid = !GUARD || (n0 + cr) < N;

  unsigned aL[2], bL[2];
  aL[0] = ldsAddr(&As[0][cr * 64 + ch * 32]);
  aL[1] = ldsAddr(&As[1][cr * 64 + ch * 32]);
  bL[0] = ldsAddr(&Bs[0][cr * 64 + ch * 32]);
  bL[1] = ldsAddr(&Bs[1][cr * 64 + ch * 32]);

  auto async_copy = [&](int p, int k0) {
#pragma unroll
    for (int i = 0; i < 4; ++i) {
      asm volatile("global_load_async_to_lds_b128 %0, %1, off"
                   :: "v"(aL[p] + 16u * i), "v"(aptr + k0 + 8 * i) : "memory");
    }
    if (bvalid) {
#pragma unroll
      for (int i = 0; i < 4; ++i) {
        asm volatile("global_load_async_to_lds_b128 %0, %1, off"
                     :: "v"(bL[p] + 16u * i), "v"(bptr + k0 + 8 * i) : "memory");
      }
    }
  };

  if (GUARD && !bvalid) {   // zero rows stay zero for every K-tile
    uint4 z = {0, 0, 0, 0};
#pragma unroll
    for (int i = 0; i < 4; ++i) {
      *(uint4*)(&Bs[0][cr * 64 + ch * 32 + 8 * i]) = z;
      *(uint4*)(&Bs[1][cr * 64 + ch * 32 + 8 * i]) = z;
    }
  }

  f32x8 acc[2][4] = {};

  async_copy(0, 0);
  asm volatile("s_wait_asynccnt 0x0" ::: "memory");
  __syncthreads();

  int p = 0;
  for (int k0 = 0; k0 < K; k0 += 64) {
    if (k0 + 64 < K) async_copy(p ^ 1, k0 + 64);   // overlaps WMMAs below
    if (k0 + 128 < K) {                             // gfx1250 global_prefetch_b8
      __builtin_prefetch(aptr + k0 + 128, 0, 1);
      __builtin_prefetch(bptr + k0 + 128, 0, 1);
    }

    const u16* asb = As[p];
    const u16* bsb = Bs[p];
#pragma unroll
    for (int ks = 0; ks < 64; ks += 32) {
      AFrag af[2], bfr[4];
#pragma unroll
      for (int mi = 0; mi < 2; ++mi) {
        int mrow = wm * 32 + mi * 16 + l16;
#pragma unroll
        for (int gg = 0; gg < 8; ++gg) {
          int k = ks + ((gg & 4) ? 16 : 0) + half * 8 + (gg & 3) * 2;
          af[mi].u[gg] = *(const unsigned int*)(&asb[mrow * 64 + k]);
        }
      }
#pragma unroll
      for (int ni = 0; ni < 4; ++ni) {
        int ncol = wn * 64 + ni * 16 + l16;
#pragma unroll
        for (int gg = 0; gg < 8; ++gg) {
          int k = ks + ((gg & 4) ? 16 : 0) + half * 8 + (gg & 3) * 2;
          bfr[ni].u[gg] = *(const unsigned int*)(&bsb[ncol * 64 + k]);
        }
      }
#pragma unroll
      for (int mi = 0; mi < 2; ++mi)
#pragma unroll
        for (int ni = 0; ni < 4; ++ni)
          acc[mi][ni] = __builtin_amdgcn_wmma_f32_16x16x32_bf16(
              false, af[mi].v, false, bfr[ni].v, (short)0, acc[mi][ni], false, false);
    }

    asm volatile("s_wait_asynccnt 0x0" ::: "memory");  // next buffer filled
    __syncthreads();                                    // all waves' copies visible
    p ^= 1;
  }

  // epilogue: lane holds column (n), 8 rows per accumulator (VGPR v -> M = v + 8*half)
#pragma unroll
  for (int mi = 0; mi < 2; ++mi) {
#pragma unroll
    for (int ni = 0; ni < 4; ++ni) {
      int col = n0 + wn * 64 + ni * 16 + l16;
      if (!GUARD || col < N) {
        float bv = 0.f;
        if constexpr (EPI == 2 || EPI == 4 || EPI == 5) bv = bias[col];
#pragma unroll
        for (int v = 0; v < 8; ++v) {
          long row = (long)m0 + wm * 32 + mi * 16 + half * 8 + v;
          float x = acc[mi][ni][v];
          if constexpr (EPI == 0) {
            C0[row * (long)N + col] = x;
          } else if constexpr (EPI == 1) {
            if (col < splitN) C0[row * (long)splitN + col] = x;
            else              C1[row * (long)splitN + (col - splitN)] = f2bf(x);
          } else if constexpr (EPI == 2) {
            x += bv;
            x = (x > 20.f) ? x : __logf(1.f + __expf(x));
            C0[row * (long)N + col] = x;
          } else if constexpr (EPI == 3) {
            C1[row * (long)N + col] = f2bf(x);
          } else if constexpr (EPI == 4) {
            x += bv;
            x = (x > 0.f) ? x : 0.01f * x;
            C1[row * (long)N + col] = f2bf(x);
          } else {  // EPI 5
            C0[row * (long)N + col] = x + bv;
          }
        }
      }
    }
  }
}

// ---------------- host launcher ----------------
static inline int g1(long n) { return (int)((n + 255) / 256); }

extern "C" void kernel_launch(void* const* d_in, const int* in_sizes, int n_in,
                              void* d_out, int out_size, void* d_ws, size_t ws_size,
                              hipStream_t stream) {
  const float* x      = (const float*)d_in[0];
  const float* W_in   = (const float*)d_in[1];
  const float* conv_w = (const float*)d_in[2];
  const float* conv_b = (const float*)d_in[3];
  const float* W_xp   = (const float*)d_in[4];
  const float* W_dt   = (const float*)d_in[5];
  const float* b_dt   = (const float*)d_in[6];
  const float* A_log  = (const float*)d_in[7];
  const float* Dvec   = (const float*)d_in[8];
  const float* W_out  = (const float*)d_in[9];
  const float* W_ff1  = (const float*)d_in[10];
  const float* b_ff1  = (const float*)d_in[11];
  const float* W_ff2  = (const float*)d_in[12];
  const float* b_ff2  = (const float*)d_in[13];
  float* out = (float*)d_out;

  char* ws = (char*)d_ws;
  size_t off = 0;
  auto alloc = [&](size_t bytes) { size_t r = off; off += (bytes + 255) & ~(size_t)255; return r; };

  const long NT = NTOK;
  // all weight copies are bf16, pre-transposed to [N][K]
  u16*   Winb  = (u16*)  (ws + alloc((size_t)DMODEL * 2 * DINNER * 2));  // [3072][768]
  u16*   Wxpb  = (u16*)  (ws + alloc((size_t)112 * DINNER * 2));         // [112][1536]
  u16*   Wdtb  = (u16*)  (ws + alloc((size_t)DINNER * 64 * 2));          // [1536][64]
  u16*   Woutb = (u16*)  (ws + alloc((size_t)DMODEL * DINNER * 2));      // [768][1536]
  u16*   Wff1b = (u16*)  (ws + alloc((size_t)NHID * DMODEL * 2));        // [2048][768]
  u16*   Wff2b = (u16*)  (ws + alloc((size_t)ABINS * NHID * 2));         // [256][2048]
  u16*   Xbf   = (u16*)  (ws + alloc((size_t)NT * DMODEL * 2));   // reused as mo_bf
  float* xcraw = (float*)(ws + alloc((size_t)NT * DINNER * 4));   // reused as yscan
  u16*   zbf   = (u16*)  (ws + alloc((size_t)NT * DINNER * 2));
  float* xcs   = (float*)(ws + alloc((size_t)NT * DINNER * 4));
  u16*   xcsbf = (u16*)  (ws + alloc((size_t)NT * DINNER * 2));   // reused as ygate_bf
  float* proj  = (float*)(ws + alloc((size_t)NT * 112 * 4));
  u16*   dtin  = (u16*)  (ws + alloc((size_t)NT * 64 * 2));
  float* dtf   = (float*)(ws + alloc((size_t)NT * DINNER * 4));   // reused as h1_bf
  float* yscan = xcraw;
  u16*   ygb   = xcsbf;
  u16*   mob   = Xbf;
  u16*   h1b   = (u16*)dtf;

  // 1) bf16 conversions; weights transposed to [N][K] (fit in L2; per call)
  k_f2bf_t<<<g1((long)DMODEL * 2 * DINNER), 256, 0, stream>>>(W_in,  Winb,  DMODEL, 2 * DINNER);
  k_f2bf_t<<<g1((long)DINNER * 112),        256, 0, stream>>>(W_xp,  Wxpb,  DINNER, 112);
  k_pad_wdt_t<<<g1((long)DINNER * 64),      256, 0, stream>>>(W_dt,  Wdtb);
  k_f2bf_t<<<g1((long)DINNER * DMODEL),     256, 0, stream>>>(W_out, Woutb, DINNER, DMODEL);
  k_f2bf_t<<<g1((long)DMODEL * NHID),       256, 0, stream>>>(W_ff1, Wff1b, DMODEL, NHID);
  k_f2bf_t<<<g1((long)NHID * ABINS),        256, 0, stream>>>(W_ff2, Wff2b, NHID, ABINS);
  k_f2bf<<<g1(NT * DMODEL),                 256, 0, stream>>>(x,     Xbf,   NT * DMODEL);

  const dim3 blk(256);
  const int MY = NTOK / 128;  // 256

  // 2) xz = x @ W_in  (split xc f32 / z bf16)
  gemm_bf16<1, false><<<dim3((2 * DINNER) / 128, MY), blk, 0, stream>>>(
      Xbf, Winb, NTOK, 2 * DINNER, DMODEL, DMODEL, DMODEL, xcraw, zbf, nullptr, DINNER);

  // 3) causal depthwise conv + SiLU
  k_conv<<<g1(NT * DINNER), 256, 0, stream>>>(xcraw, conv_w, conv_b, xcs, xcsbf);

  // 4) proj = xcs @ W_xproj (N=112, edge-guarded instantiation only)
  gemm_bf16<0, true><<<dim3(1, MY), blk, 0, stream>>>(
      xcsbf, Wxpb, NTOK, 112, DINNER, DINNER, DINNER, proj, nullptr, nullptr, 0);

  // 5) dt_in (pad K 48->64), dt = softplus(dt_in @ W_dt + b_dt)
  k_extract_dt<<<g1(NT * 64), 256, 0, stream>>>(proj, dtin);
  gemm_bf16<2, false><<<dim3(DINNER / 128, MY), blk, 0, stream>>>(
      dtin, Wdtb, NTOK, DINNER, 64, 64, 64, dtf, nullptr, b_dt, 0);

  // 6) selective scan (recurrent over L; 4 lanes x 8 states per channel)
  k_scan<<<(NBATCH * DINNER * 4) / 256, 256, 0, stream>>>(dtf, proj, xcs, A_log, yscan);

  // 7) gate: (y + xcs*D) * silu(z) -> bf16
  k_gate<<<g1(NT * DINNER), 256, 0, stream>>>(yscan, xcs, Dvec, zbf, ygb);

  // 8) mamba_out = y_g @ W_out -> bf16
  gemm_bf16<3, false><<<dim3(DMODEL / 128, MY), blk, 0, stream>>>(
      ygb, Woutb, NTOK, DMODEL, DINNER, DINNER, DINNER, nullptr, mob, nullptr, 0);

  // 9) h1 = leaky_relu(mamba_out @ W_ff1 + b_ff1) -> bf16
  gemm_bf16<4, false><<<dim3(NHID / 128, MY), blk, 0, stream>>>(
      mob, Wff1b, NTOK, NHID, DMODEL, DMODEL, DMODEL, nullptr, h1b, b_ff1, 0);

  // 10) out = h1 @ W_ff2 + b_ff2 -> f32 d_out
  gemm_bf16<5, false><<<dim3(ABINS / 128, MY), blk, 0, stream>>>(
      h1b, Wff2b, NTOK, ABINS, NHID, NHID, NHID, out, nullptr, b_ff2, 0);
}
